// MultiHeadAttention_61692910240003
// MI455X (gfx1250) — matrix operand order repaired and checked by reading, not compiled, not run on hardware
//
#include <hip/hip_runtime.h>
#include <stdint.h>

// ---- problem sizes (fixed by the reference) ----
static constexpr int BN = 4;      // batch
static constexpr int SS = 2048;   // sequence
static constexpr int EE = 1024;   // embed
static constexpr int HN = 16;     // heads
static constexpr int DH = 64;     // head dim
static constexpr int STRIDE = 2052;         // padded LDS row stride (floats)
static constexpr float LOG2E = 1.4426950408889634f;

typedef __attribute__((ext_vector_type(16))) __bf16 v16bf;
typedef __attribute__((ext_vector_type(8)))  __bf16 v8bf;
typedef __attribute__((ext_vector_type(8)))  float  v8f;

union ABfrag { v16bf v; v8bf h[2]; };

static __device__ inline __bf16 f2bf(float f) {
    union { float f; uint32_t u; } x; x.f = f;
    uint32_t r = (x.u + 0x7fffu + ((x.u >> 16) & 1u)) >> 16;
    union { unsigned short s; __bf16 b; } y; y.s = (unsigned short)r;
    return y.b;
}

static __device__ inline float fast_exp2(float x) {
    return __builtin_amdgcn_exp2f(x);   // v_exp_f32 (TRANS, co-executes with WMMA)
}

// ---------------- fp32 -> bf16 convert ----------------
__global__ __launch_bounds__(256) void cvt_f32_bf16(const float* __restrict__ src,
                                                    __bf16* __restrict__ dst, int n) {
    int i = blockIdx.x * blockDim.x + threadIdx.x;
    int stride = gridDim.x * blockDim.x;
    for (; i < n; i += stride) dst[i] = f2bf(src[i]);
}

// ---------------- QKV projection GEMM: out = X @ W^T + b ----------------
// block = 8 waves, block tile 64(M) x 128(N); wave tile 32x32 (2x2 WMMA accs)
__global__ __launch_bounds__(256) void qkv_gemm(
    const __bf16* __restrict__ Xb, const __bf16* __restrict__ Wall,
    const float* __restrict__ biasq, const float* __restrict__ biask,
    const float* __restrict__ biasv,
    __bf16* __restrict__ Qb, __bf16* __restrict__ Kb, __bf16* __restrict__ Vt) {
    const int tid = threadIdx.x;
    const int lane = tid & 31, w = tid >> 5;
    const int hl = lane >> 4, l16 = lane & 15;
    const int m0 = blockIdx.x * 64, n0 = blockIdx.y * 128;
    const int proj = blockIdx.z;
    const __bf16* Wm = Wall + (size_t)proj * EE * EE;
    const float* bias = (proj == 0) ? biasq : ((proj == 1) ? biask : biasv);

    const int mbase = m0 + (w & 1) * 32;
    const int nbase = n0 + (w >> 1) * 32;

    v8f acc[2][2] = { { {}, {} }, { {}, {} } };

    for (int kc = 0; kc < EE; kc += 32) {
        ABfrag afr[2], bfr[2];
#pragma unroll
        for (int am = 0; am < 2; ++am) {
            const __bf16* p = Xb + (size_t)(mbase + am * 16 + l16) * EE + kc + 8 * hl;
            afr[am].h[0] = *(const v8bf*)(p);
            afr[am].h[1] = *(const v8bf*)(p + 16);
        }
#pragma unroll
        for (int bn = 0; bn < 2; ++bn) {
            const __bf16* p = Wm + (size_t)(nbase + bn * 16 + l16) * EE + kc + 8 * hl;
            bfr[bn].h[0] = *(const v8bf*)(p);
            bfr[bn].h[1] = *(const v8bf*)(p + 16);
        }
#pragma unroll
        for (int am = 0; am < 2; ++am)
#pragma unroll
            for (int bn = 0; bn < 2; ++bn)
                acc[am][bn] = __builtin_amdgcn_wmma_f32_16x16x32_bf16(
                    false, afr[am].v, false, bfr[bn].v, (short)0, acc[am][bn],
                    false, false);
    }

#pragma unroll
    for (int am = 0; am < 2; ++am)
#pragma unroll
        for (int bn = 0; bn < 2; ++bn)
#pragma unroll
            for (int j = 0; j < 8; ++j) {
                int gm = mbase + am * 16 + j + 8 * hl;       // row in [B*S]
                int gn = nbase + bn * 16 + l16;              // col in [E]
                float v = acc[am][bn][j] + bias[gn];
                int b = gm >> 11, s = gm & (SS - 1);
                int hh = gn >> 6, d = gn & (DH - 1);
                __bf16 bf = f2bf(v);
                size_t bhs = (size_t)(b * HN + hh);
                if (proj == 0)      Qb[(bhs * SS + s) * DH + d] = bf;    // [B,H,S,D]
                else if (proj == 1) Kb[(bhs * SS + s) * DH + d] = bf;    // [B,H,S,D]
                else                Vt[(bhs * DH + d) * SS + s] = bf;    // [B,H,D,S]
            }
}

// ---------------- attention: one block per (b, h, 32 K-rows) ----------------
__global__ __launch_bounds__(256) void attn_kernel(
    const __bf16* __restrict__ Qb, const __bf16* __restrict__ Kb,
    const __bf16* __restrict__ Vt, const int* __restrict__ xraw,
    float* __restrict__ out) {
    extern __shared__ __align__(16) char smem_raw[];
    float* strip   = (float*)smem_raw;            // [32][STRIDE]
    float* partial = strip + 32 * STRIDE;         // [32][8]
    float* rowmax  = partial + 32 * 8;            // [32]
    float* rowsum  = rowmax + 32;                 // [32]

    const int tid = threadIdx.x;
    const int lane = tid & 31, w = tid >> 5;
    const int hl = lane >> 4, l16 = lane & 15;
    const int k0 = blockIdx.x * 32;
    const int b = blockIdx.z;
    const size_t bh = (size_t)(b * HN + blockIdx.y);
    const __bf16* Qh = Qb + bh * SS * DH;
    const __bf16* Kh = Kb + bh * SS * DH;
    const __bf16* Vh = Vt + bh * DH * SS;

    const int mh = w & 1;       // which 16-row half of the 32-row tile
    const int qsw = w >> 1;     // q-stream 0..3

    // ---- Phase A: score strip [32 x 2048] into LDS ----
    ABfrag aK[2];
    {
        const __bf16* p = Kh + (size_t)(k0 + mh * 16 + l16) * DH;
#pragma unroll
        for (int c = 0; c < 2; ++c) {
            aK[c].h[0] = *(const v8bf*)(p + c * 32 + 8 * hl);
            aK[c].h[1] = *(const v8bf*)(p + c * 32 + 16 + 8 * hl);
        }
    }
    const float scale = 0.03125f;   // 1/sqrt(E) = 1/32
    for (int t = 0; t < 32; ++t) {
        int q0 = (qsw + 4 * t) * 16;
        ABfrag bQ[2];
        const __bf16* p = Qh + (size_t)(q0 + l16) * DH;
#pragma unroll
        for (int c = 0; c < 2; ++c) {
            bQ[c].h[0] = *(const v8bf*)(p + c * 32 + 8 * hl);
            bQ[c].h[1] = *(const v8bf*)(p + c * 32 + 16 + 8 * hl);
        }
        v8f acc = {};
        acc = __builtin_amdgcn_wmma_f32_16x16x32_bf16(false, aK[0].v, false, bQ[0].v,
                                                      (short)0, acc, false, false);
        acc = __builtin_amdgcn_wmma_f32_16x16x32_bf16(false, aK[1].v, false, bQ[1].v,
                                                      (short)0, acc, false, false);
        int q = q0 + l16;
        float mv = (xraw[b * SS + q] != 0) ? 0.0f : -1e20f;  // query-axis mask
#pragma unroll
        for (int j = 0; j < 8; ++j)
            strip[(mh * 16 + j + 8 * hl) * STRIDE + q] = acc[j] * scale + mv;
    }
    __syncthreads();

    // ---- Phase B: per-row max and sum-of-exp over q ----
    {
        int rr = tid >> 3, p = tid & 7;
        const float* sp = strip + rr * STRIDE + p * 256;
        float m = -3.0e38f;
        for (int q = 0; q < 256; q += 4) {
            float4 v4 = *(const float4*)(sp + q);
            m = fmaxf(m, fmaxf(fmaxf(v4.x, v4.y), fmaxf(v4.z, v4.w)));
        }
        partial[rr * 8 + p] = m;
    }
    __syncthreads();
    if (tid < 32) {
        float m = partial[tid * 8];
#pragma unroll
        for (int i = 1; i < 8; ++i) m = fmaxf(m, partial[tid * 8 + i]);
        rowmax[tid] = m;
    }
    __syncthreads();
    {
        int rr = tid >> 3, p = tid & 7;
        float rm = rowmax[rr];
        const float* sp = strip + rr * STRIDE + p * 256;
        float s = 0.0f;
        for (int q = 0; q < 256; q += 4) {
            float4 v4 = *(const float4*)(sp + q);
            s += fast_exp2((v4.x - rm) * LOG2E) + fast_exp2((v4.y - rm) * LOG2E) +
                 fast_exp2((v4.z - rm) * LOG2E) + fast_exp2((v4.w - rm) * LOG2E);
        }
        partial[rr * 8 + p] = s;
    }
    __syncthreads();
    if (tid < 32) {
        float s = 0.0f;
#pragma unroll
        for (int i = 0; i < 8; ++i) s += partial[tid * 8 + i];
        rowsum[tid] = s;
    }
    __syncthreads();

    // ---- Phase C: ctx = exp(P) @ V via WMMA, q-streamed per wave ----
    const int r = mh * 16 + l16;            // lane's A-fragment row
    const float rmL = rowmax[r];
    v8f ctx[4] = { {}, {}, {}, {} };
    for (int t = 0; t < 16; ++t) {
        int qb = (qsw + 4 * t) * 32;
        const float* sp = strip + r * STRIDE + qb;
        float pv[16];
        *(float4*)(pv + 0)  = *(const float4*)(sp + 8 * hl);
        *(float4*)(pv + 4)  = *(const float4*)(sp + 8 * hl + 4);
        *(float4*)(pv + 8)  = *(const float4*)(sp + 16 + 8 * hl);
        *(float4*)(pv + 12) = *(const float4*)(sp + 16 + 8 * hl + 4);
        ABfrag pa;
#pragma unroll
        for (int i = 0; i < 16; ++i)
            pa.v[i] = f2bf(fast_exp2((pv[i] - rmL) * LOG2E));
#pragma unroll
        for (int dt = 0; dt < 4; ++dt) {
            ABfrag bV;
            const __bf16* vp = Vh + (size_t)(dt * 16 + l16) * SS + qb + 8 * hl;
            bV.h[0] = *(const v8bf*)(vp);
            bV.h[1] = *(const v8bf*)(vp + 16);
            ctx[dt] = __builtin_amdgcn_wmma_f32_16x16x32_bf16(
                false, pa.v, false, bV.v, (short)0, ctx[dt], false, false);
        }
    }
    __syncthreads();
    // partial ctx -> LDS (reuse strip): [row 32][qs 4][d 64]
#pragma unroll
    for (int dt = 0; dt < 4; ++dt)
#pragma unroll
        for (int j = 0; j < 8; ++j) {
            int row = mh * 16 + j + 8 * hl;
            strip[(row * 4 + qsw) * 64 + dt * 16 + l16] = ctx[dt][j];
        }
    __syncthreads();
    // reduce 4 q-streams, divide by rowsum, write (reference's flat [B,H,S,D])
#pragma unroll
    for (int u = 0; u < 8; ++u) {
        int e = tid * 8 + u;
        int row = e >> 6, d = e & 63;
        const float* pp = strip + (row * 4) * 64 + d;
        float sum = pp[0] + pp[64] + pp[128] + pp[192];
        out[(bh * SS + (k0 + row)) * DH + d] = sum / rowsum[row];
    }
}

extern "C" void kernel_launch(void* const* d_in, const int* in_sizes, int n_in,
                              void* d_out, int out_size, void* d_ws, size_t ws_size,
                              hipStream_t stream) {
    const float* x  = (const float*)d_in[0];
    const int*   xr = (const int*)d_in[1];
    const float* Wq = (const float*)d_in[2];
    const float* bq = (const float*)d_in[3];
    const float* Wk = (const float*)d_in[4];
    const float* bk = (const float*)d_in[5];
    const float* Wv = (const float*)d_in[6];
    const float* bv = (const float*)d_in[7];
    float* out = (float*)d_out;

    const size_t nX = (size_t)BN * SS * EE;      // 8.4M
    const size_t nW = (size_t)EE * EE;           // 1M
    const size_t nH = (size_t)BN * HN * SS * DH; // 8.4M

    __bf16* Xb   = (__bf16*)d_ws;
    __bf16* Wall = Xb + nX;
    __bf16* Qb   = Wall + 3 * nW;
    __bf16* Kb   = Qb + nH;
    __bf16* Vt   = Kb + nH;

    cvt_f32_bf16<<<4096, 256, 0, stream>>>(x, Xb, (int)nX);
    cvt_f32_bf16<<<1024, 256, 0, stream>>>(Wq, Wall + 0 * nW, (int)nW);
    cvt_f32_bf16<<<1024, 256, 0, stream>>>(Wk, Wall + 1 * nW, (int)nW);
    cvt_f32_bf16<<<1024, 256, 0, stream>>>(Wv, Wall + 2 * nW, (int)nW);

    dim3 g1((BN * SS) / 64, EE / 128, 3);
    qkv_gemm<<<g1, 256, 0, stream>>>(Xb, Wall, bq, bk, bv, Qb, Kb, Vt);

    size_t shmem = (size_t)(32 * STRIDE + 32 * 8 + 32 + 32) * sizeof(float); // ~264 KB
    (void)hipFuncSetAttribute((const void*)attn_kernel,
                              hipFuncAttributeMaxDynamicSharedMemorySize, (int)shmem);
    dim3 g2(SS / 32, HN, BN);
    attn_kernel<<<g2, 256, shmem, stream>>>(Qb, Kb, Vt, xr, out);
}
